// MultiHeadAttention_300647711299
// MI455X (gfx1250) — matrix operand order repaired and checked by reading, not compiled, not run on hardware
//
#include <hip/hip_runtime.h>

// ---------------------------------------------------------------------------
// MHA forward for MI455X (gfx1250): bf16 WMMA + TDM (tensor_load_to_lds)
// double-buffered GEMMs + flash attention with DPP16 row reductions.
// D_MODEL=1024, H=16, d_k=64, B=4, S=2048, M=B*S=8192.
// ---------------------------------------------------------------------------

typedef __attribute__((ext_vector_type(16))) __bf16 bf16x16;
typedef __attribute__((ext_vector_type(8)))  __bf16 bf16x8;
typedef __attribute__((ext_vector_type(4)))  __bf16 bf16x4;
typedef __attribute__((ext_vector_type(8)))  float  f32x8;
typedef __attribute__((ext_vector_type(4)))  unsigned u32x4;
typedef __attribute__((ext_vector_type(8)))  int      i32x8;
typedef __attribute__((ext_vector_type(4)))  int      i32x4;

#define DEV __device__ __forceinline__

#if defined(__has_builtin)
#if __has_builtin(__builtin_amdgcn_tensor_load_to_lds) && \
    __has_builtin(__builtin_amdgcn_s_wait_tensorcnt)
#define USE_TDM 1
#endif
#endif

DEV __bf16 f2bf(float x) {
    union { float f; unsigned u; } cv; cv.f = x;
    unsigned r = (cv.u + 0x7FFFu + ((cv.u >> 16) & 1u)) >> 16;   // RNE
    union { unsigned short s; __bf16 b; } cb; cb.s = (unsigned short)r;
    return cb.b;
}

DEV bf16x16 frag_cat(bf16x8 lo, bf16x8 hi) {
    return __builtin_shufflevector(lo, hi, 0,1,2,3,4,5,6,7,8,9,10,11,12,13,14,15);
}

// A-operand fragment (16x32 bf16), row-major source, leading dim ld (ISA 7.12.2):
// lane<16 -> row row0+lane,      K {k0..k0+7, k0+16..k0+23}
// lane>=16 -> row row0+lane-16,  K {k0+8..15, k0+24..31}
DEV bf16x16 load_frag_a(const __bf16* base, int ld, int row0, int k0, int lane) {
    const __bf16* p = base + (size_t)(row0 + (lane & 15)) * ld + k0 + ((lane >> 4) << 3);
    return frag_cat(*(const bf16x8*)p, *(const bf16x8*)(p + 16));
}

// B-operand fragment (32x16): B[k][n] = S[col0+n][k], S row-major, ld:
// lane<16 -> col lane, K {k0..k0+7, k0+16..k0+23}; lane>=16 -> col lane-16, K {k0+8..15, k0+24..31}
DEV bf16x16 load_frag_b(const __bf16* base, int ld, int col0, int k0, int lane) {
    const __bf16* p = base + (size_t)(col0 + (lane & 15)) * ld + k0 + ((lane >> 4) << 3);
    return frag_cat(*(const bf16x8*)p, *(const bf16x8*)(p + 16));
}

DEV f32x8 wmma_bf16(bf16x16 a, bf16x16 b, f32x8 c) {
    return __builtin_amdgcn_wmma_f32_16x16x32_bf16(false, a, false, b,
                                                   (short)0, c, false, false);
}

// ---- DPP16 16-lane butterfly reductions (LDS-free, co-executes with WMMA) ----
template <int CTRL>
DEV float dpp_xfer(float x) {
    union { float f; int i; } a, r;
    a.f = x;
    r.i = __builtin_amdgcn_update_dpp(a.i, a.i, CTRL, 0xF, 0xF, false);
    return r.f;
}
DEV float rowmax16(float x) {
    x = fmaxf(x, dpp_xfer<0xB1>(x));    // quad_perm(1,0,3,2) : lane^1
    x = fmaxf(x, dpp_xfer<0x4E>(x));    // quad_perm(2,3,0,1) : lane^2
    x = fmaxf(x, dpp_xfer<0x141>(x));   // row_half_mirror    : lane^7
    x = fmaxf(x, dpp_xfer<0x140>(x));   // row_mirror         : lane^15
    return x;
}
DEV float rowsum16(float x) {
    x += dpp_xfer<0xB1>(x);
    x += dpp_xfer<0x4E>(x);
    x += dpp_xfer<0x141>(x);
    x += dpp_xfer<0x140>(x);
    return x;
}

// ---------------------------------------------------------------------------
// TDM: 2D tile (tile_d1 rows x tile_d0 elems of 2B) global -> LDS, with
// native LDS row padding: 128B data rows + 16B pad -> 144B LDS row stride.
// D# per CDNA5 ISA. Groups 2/3 zero (2D tensor);
// this toolchain's builtin is the 6-operand form (extra 8-dword group = 0).
// ---------------------------------------------------------------------------
#ifdef USE_TDM
DEV void tdm_load_2d(unsigned lds_byte_off, const void* gsrc,
                     unsigned tensor_d0, unsigned tensor_d1,
                     unsigned tile_d0, unsigned tile_d1, unsigned stride0) {
    unsigned long long ga = (unsigned long long)gsrc;
    u32x4 g0;
    g0[0] = 1u;                                   // count=1 (valid user D#)
    g0[1] = lds_byte_off;                         // lds_addr
    g0[2] = (unsigned)ga;                         // global_addr[31:0]
    g0[3] = (unsigned)((ga >> 32) & 0x01FFFFFFu)  // global_addr[56:32]
          | (2u << 30);                           // type=2 ("image")
    i32x8 g1;
    g1[0] = (int)((1u << 16)                      // data_size = 2B
                | (1u << 20)                      // pad_enable
                | (4u << 22)                      // pad_interval: 32 DWORDs (128B)
                | (3u << 25));                    // pad_amount:   4 DWORDs (16B)
    g1[1] = (int)((tensor_d0 & 0xFFFFu) << 16);                       // dim0[15:0]
    g1[2] = (int)((tensor_d0 >> 16) | ((tensor_d1 & 0xFFFFu) << 16)); // dim0 hi, dim1 lo
    g1[3] = (int)((tensor_d1 >> 16) | (tile_d0 << 16));               // dim1 hi, tile0
    g1[4] = (int)(tile_d1);                                           // tile1, tile2=0
    g1[5] = (int)stride0;                                             // dim0_stride lo
    g1[6] = 0;                                                        // stride hi, dim1_stride=0
    g1[7] = 0;
    i32x4 z4 = {0, 0, 0, 0};
    i32x8 z8 = {0, 0, 0, 0, 0, 0, 0, 0};
    __builtin_amdgcn_tensor_load_to_lds(g0, g1, z4, z4, z8, 0);
}
#endif

// ---------------------------------------------------------------------------
// fp32 -> bf16 conversion, 4 elements / thread
// ---------------------------------------------------------------------------
__global__ __launch_bounds__(256)
void f32_to_bf16_vec4(const float* __restrict__ in, __bf16* __restrict__ out, int n4) {
    int i = blockIdx.x * 256 + threadIdx.x;
    if (i < n4) {
        float4 v = ((const float4*)in)[i];
        bf16x4 o;
        o[0] = f2bf(v.x); o[1] = f2bf(v.y); o[2] = f2bf(v.z); o[3] = f2bf(v.w);
        ((bf16x4*)out)[i] = o;
    }
}

// ---------------------------------------------------------------------------
// GEMM: C[M=8192, N=1024] = A @ W^T + bias, K=1024.
// Block tile 128x128, 8 waves (4x2), wave tile 32x64.
// K staged 64 at a time through LDS by TDM (wave 0 issues one DMA per tile),
// double-buffered: DMA of stage k+1 overlaps the 16 WMMAs of stage k.
// LDS: 4 regions (2 bufs x {A,B}) of 128 rows x 72 bf16 (144B padded rows).
// MODE: 0=Q ((v+b)/8 -> [B,H,S,dk] bf16), 1=K (-> [B,H,S,dk] bf16),
//       2=V (-> [B,H,dk,S] bf16, transposed), 3=O (-> [M,1024] fp32)
// ---------------------------------------------------------------------------
#define T_LD   72              // padded LDS row stride (elements)
#define T_ELEM (128 * T_LD)    // elements per tile region

template <int MODE>
__global__ __launch_bounds__(256)
void gemm_kernel(const __bf16* __restrict__ A, const __bf16* __restrict__ W,
                 const float* __restrict__ bias, void* __restrict__ out) {
    __shared__ __bf16 lds[4 * T_ELEM];           // 73728 B, assumed LDS offset 0
    constexpr int KD = 1024;
    const int tid  = threadIdx.x;
    const int lane = tid & 31;
    const int wave = tid >> 5;
    const int rowB = blockIdx.x * 128;           // block row base (M)
    const int colB = blockIdx.y * 128;           // block col base (N)
    const int rowL = (wave & 3) * 32;            // wave-local row base
    const int colL = (wave >> 2) * 64;           // wave-local col base

    auto stage = [&](int buf, int k0) {
#ifdef USE_TDM
        if (wave == 0) {
            tdm_load_2d((unsigned)((2 * buf + 0) * T_ELEM * 2),
                        A + (size_t)rowB * KD + k0, KD, 8192, 64, 128, KD);
            tdm_load_2d((unsigned)((2 * buf + 1) * T_ELEM * 2),
                        W + (size_t)colB * KD + k0, KD, 1024, 64, 128, KD);
        }
#else
        __bf16* la = &lds[(2 * buf + 0) * T_ELEM];
        __bf16* lb = &lds[(2 * buf + 1) * T_ELEM];
        #pragma unroll
        for (int i = 0; i < 4; ++i) {            // 1024 b128 chunks per tile
            int c = tid + i * 256;
            int r = c >> 2, ch = (c & 3) << 3;
            *(bf16x8*)(la + r * T_LD + ch) = *(const bf16x8*)(A + (size_t)(rowB + r) * KD + k0 + ch);
            *(bf16x8*)(lb + r * T_LD + ch) = *(const bf16x8*)(W + (size_t)(colB + r) * KD + k0 + ch);
        }
#endif
    };

    f32x8 acc[2][4] = {};

    stage(0, 0);
#ifdef USE_TDM
    __builtin_amdgcn_s_wait_tensorcnt(0);
#endif
    __syncthreads();

    for (int ks = 0; ks < 16; ++ks) {
        const int buf = ks & 1;
        if (ks + 1 < 16) stage(buf ^ 1, (ks + 1) * 64);

        const __bf16* Ab = &lds[(2 * buf + 0) * T_ELEM];
        const __bf16* Bb = &lds[(2 * buf + 1) * T_ELEM];
        #pragma unroll
        for (int kk = 0; kk < 64; kk += 32) {
            bf16x16 a0 = load_frag_a(Ab, T_LD, rowL,      kk, lane);
            bf16x16 a1 = load_frag_a(Ab, T_LD, rowL + 16, kk, lane);
            bf16x16 b0 = load_frag_b(Bb, T_LD, colL,      kk, lane);
            bf16x16 b1 = load_frag_b(Bb, T_LD, colL + 16, kk, lane);
            bf16x16 b2 = load_frag_b(Bb, T_LD, colL + 32, kk, lane);
            bf16x16 b3 = load_frag_b(Bb, T_LD, colL + 48, kk, lane);
            acc[0][0] = wmma_bf16(a0, b0, acc[0][0]);
            acc[0][1] = wmma_bf16(a0, b1, acc[0][1]);
            acc[0][2] = wmma_bf16(a0, b2, acc[0][2]);
            acc[0][3] = wmma_bf16(a0, b3, acc[0][3]);
            acc[1][0] = wmma_bf16(a1, b0, acc[1][0]);
            acc[1][1] = wmma_bf16(a1, b1, acc[1][1]);
            acc[1][2] = wmma_bf16(a1, b2, acc[1][2]);
            acc[1][3] = wmma_bf16(a1, b3, acc[1][3]);
        }
#ifdef USE_TDM
        __builtin_amdgcn_s_wait_tensorcnt(0);    // stage k+1 DMA landed
#endif
        __syncthreads();
    }

    // Epilogue. C/D layout: VGPR r -> rows r (lanes 0-15) / r+8 (lanes 16-31).
    const int nn = lane & 15;
    const int rb = (lane >> 4) << 3;
    const int row0 = rowB + rowL;
    const int col0 = colB + colL;
    #pragma unroll
    for (int j = 0; j < 4; ++j) {
        const int n = col0 + j * 16 + nn;
        const float bv = bias[n];
        #pragma unroll
        for (int i = 0; i < 2; ++i) {
            #pragma unroll
            for (int r = 0; r < 8; ++r) {
                const int m = row0 + i * 16 + rb + r;
                float v = acc[i][j][r] + bv;
                if (MODE == 3) {
                    ((float*)out)[(size_t)m * 1024 + n] = v;
                } else {
                    const int b = m >> 11, s = m & 2047;
                    const int h = n >> 6,  d = n & 63;
                    __bf16* o = (__bf16*)out;
                    if (MODE == 0)
                        o[(((size_t)(b * 16 + h) * 2048) + s) * 64 + d] = f2bf(v * 0.125f);
                    else if (MODE == 1)
                        o[(((size_t)(b * 16 + h) * 2048) + s) * 64 + d] = f2bf(v);
                    else // MODE == 2: V transposed [B,H,dk,S]
                        o[(((size_t)(b * 16 + h) * 64) + d) * 2048 + s] = f2bf(v);
                }
            }
        }
    }
}

// ---------------------------------------------------------------------------
// Flash attention: grid (S/128, B*H), 8 waves; each wave owns one 16-row Q
// tile (fragments register-resident) and streams 32 keys / iteration.
// Row reductions via DPP16; probabilities re-shaped C->A layout through a
// bank-conflict-padded wave-private LDS tile.
// ---------------------------------------------------------------------------
#define P_LD 40   // 16x32 bf16 P-tile padded to 40 (80B rows: 16B-aligned)

__global__ __launch_bounds__(256)
void attn_kernel(const __bf16* __restrict__ Q, const __bf16* __restrict__ K,
                 const __bf16* __restrict__ V, __bf16* __restrict__ ctx) {
    __shared__ __bf16 pbuf[8][16 * P_LD];

    const int lane = threadIdx.x & 31;
    const int wave = threadIdx.x >> 5;
    const int bh = blockIdx.y;               // b*16 + h
    const int b = bh >> 4, h = bh & 15;
    const int qrow0 = blockIdx.x * 128 + wave * 16;

    const __bf16* Qbh = Q + (size_t)bh * 2048 * 64;
    const __bf16* Kbh = K + (size_t)bh * 2048 * 64;
    const __bf16* Vbh = V + (size_t)bh * 64 * 2048;

    // Q fragments (16 rows x 64 d) resident for the whole pass
    bf16x16 qa0 = load_frag_a(Qbh, 64, qrow0, 0,  lane);
    bf16x16 qa1 = load_frag_a(Qbh, 64, qrow0, 32, lane);

    f32x8 acc[4] = {};
    float mrow[8], lrow[8];
    #pragma unroll
    for (int r = 0; r < 8; ++r) { mrow[r] = -1e30f; lrow[r] = 0.f; }

    __bf16* pb = &pbuf[wave][0];
    const int nn = lane & 15;
    const int rb = (lane >> 4) << 3;

    for (int j = 0; j < 2048; j += 32) {
        if (j + 32 < 2048) {   // global_prefetch_b8 for the next K/V block
            __builtin_prefetch(Kbh + (size_t)(j + 32) * 64, 0, 3);
            __builtin_prefetch(Vbh + (size_t)(j + 32), 0, 3);
        }

        // ---- scores: S[16 x 32] = Qtile @ K[j..j+31]^T (1/sqrt(dk) folded into Q)
        f32x8 s0 = {}, s1 = {};
        s0 = wmma_bf16(qa0, load_frag_b(Kbh, 64, j,      0,  lane), s0);
        s0 = wmma_bf16(qa1, load_frag_b(Kbh, 64, j,      32, lane), s0);
        s1 = wmma_bf16(qa0, load_frag_b(Kbh, 64, j + 16, 0,  lane), s1);
        s1 = wmma_bf16(qa1, load_frag_b(Kbh, 64, j + 16, 32, lane), s1);

        // ---- online softmax update (DPP16 half-wave row reductions)
        #pragma unroll
        for (int r = 0; r < 8; ++r) {
            const float mx   = rowmax16(fmaxf(s0[r], s1[r]));
            const float mnew = fmaxf(mrow[r], mx);
            const float f    = __expf(mrow[r] - mnew);
            mrow[r] = mnew;
            s0[r] = __expf(s0[r] - mnew);
            s1[r] = __expf(s1[r] - mnew);
            lrow[r] = lrow[r] * f + rowsum16(s0[r] + s1[r]);
            acc[0][r] *= f; acc[1][r] *= f; acc[2][r] *= f; acc[3][r] *= f;
        }

        // ---- P (C-layout) -> LDS (row-major bf16) -> A-layout fragment
        #pragma unroll
        for (int r = 0; r < 8; ++r) {
            pb[(rb + r) * P_LD + nn]      = f2bf(s0[r]);
            pb[(rb + r) * P_LD + 16 + nn] = f2bf(s1[r]);
        }
        bf16x16 pa = load_frag_a(pb, P_LD, 0, 0, lane);   // wave-private: ds ops in-order

        // ---- ctx += P @ V   (V transposed: B-operand contiguous per lane)
        acc[0] = wmma_bf16(pa, load_frag_b(Vbh, 2048, 0,  j, lane), acc[0]);
        acc[1] = wmma_bf16(pa, load_frag_b(Vbh, 2048, 16, j, lane), acc[1]);
        acc[2] = wmma_bf16(pa, load_frag_b(Vbh, 2048, 32, j, lane), acc[2]);
        acc[3] = wmma_bf16(pa, load_frag_b(Vbh, 2048, 48, j, lane), acc[3]);
    }

    // ---- normalize and store ctx as bf16 [B, S, 1024] at column h*64
    #pragma unroll
    for (int r = 0; r < 8; ++r) {
        const float inv = 1.0f / lrow[r];
        const int row = qrow0 + rb + r;
        #pragma unroll
        for (int t = 0; t < 4; ++t)
            ctx[((size_t)b * 2048 + row) * 1024 + h * 64 + t * 16 + nn] =
                f2bf(acc[t][r] * inv);
    }
}

// ---------------------------------------------------------------------------
// Host side
// ---------------------------------------------------------------------------
extern "C" void kernel_launch(void* const* d_in, const int* in_sizes, int n_in,
                              void* d_out, int out_size, void* d_ws, size_t ws_size,
                              hipStream_t stream) {
    const float* z  = (const float*)d_in[0];
    const float* Wq = (const float*)d_in[1];
    const float* bq = (const float*)d_in[2];
    const float* Wk = (const float*)d_in[3];
    const float* bk = (const float*)d_in[4];
    const float* Wv = (const float*)d_in[5];
    const float* bv = (const float*)d_in[6];
    const float* Wo = (const float*)d_in[7];
    const float* bo = (const float*)d_in[8];

    const size_t M = 8192;           // B*S
    const size_t D = 1024;
    char* ws = (char*)d_ws;
    size_t off = 0;
    __bf16* zb   = (__bf16*)(ws + off); off += M * D * 2;        // 16 MB
    __bf16* wqb  = (__bf16*)(ws + off); off += D * D * 2;        //  2 MB
    __bf16* wkb  = (__bf16*)(ws + off); off += D * D * 2;
    __bf16* wvb  = (__bf16*)(ws + off); off += D * D * 2;
    __bf16* wob  = (__bf16*)(ws + off); off += D * D * 2;
    __bf16* Qb   = (__bf16*)(ws + off); off += M * D * 2;        // [B,H,S,dk]
    __bf16* Kb   = (__bf16*)(ws + off); off += M * D * 2;        // [B,H,S,dk]
    __bf16* Vtb  = (__bf16*)(ws + off); off += M * D * 2;        // [B,H,dk,S]
    __bf16* ctxb = (__bf16*)(ws + off); off += M * D * 2;        // [B,S,D]

    // 1) fp32 -> bf16
    f32_to_bf16_vec4<<<(int)(M * D / 4 / 256), 256, 0, stream>>>(z,  zb,  (int)(M * D / 4));
    f32_to_bf16_vec4<<<(int)(D * D / 4 / 256), 256, 0, stream>>>(Wq, wqb, (int)(D * D / 4));
    f32_to_bf16_vec4<<<(int)(D * D / 4 / 256), 256, 0, stream>>>(Wk, wkb, (int)(D * D / 4));
    f32_to_bf16_vec4<<<(int)(D * D / 4 / 256), 256, 0, stream>>>(Wv, wvb, (int)(D * D / 4));
    f32_to_bf16_vec4<<<(int)(D * D / 4 / 256), 256, 0, stream>>>(Wo, wob, (int)(D * D / 4));

    // 2) fused QKV projections (scale folded into Q, V stored transposed)
    dim3 ggrid(64, 8);   // M/128 x N/128
    gemm_kernel<0><<<ggrid, 256, 0, stream>>>(zb, wqb, bq, (void*)Qb);
    gemm_kernel<1><<<ggrid, 256, 0, stream>>>(zb, wkb, bk, (void*)Kb);
    gemm_kernel<2><<<ggrid, 256, 0, stream>>>(zb, wvb, bv, (void*)Vtb);

    // 3) flash attention
    attn_kernel<<<dim3(16, 64), 256, 0, stream>>>(Qb, Kb, Vtb, ctxb);

    // 4) output projection (fp32 out)
    gemm_kernel<3><<<ggrid, 256, 0, stream>>>(ctxb, wob, bo, d_out);
}
